// SegmentationGTMatcher_19207093748300
// MI455X (gfx1250) — compile-verified, matrix operand order checked by reading, not run on hardware
//
#include <hip/hip_runtime.h>

typedef __attribute__((ext_vector_type(8))) int v8i;
typedef __attribute__((ext_vector_type(2))) float v2f;

#define MOVING_THRESHOLD 0.3f
#define MBOX 64

__global__ void sgm_init_counts(unsigned int* __restrict__ cnt, int n) {
    int i = blockIdx.x * blockDim.x + threadIdx.x;
    if (i < n) cnt[i] = 0u;
}

__device__ __forceinline__ unsigned box_test(float px, float py, float pz,
                                             float cx, float cy, float cz,
                                             float hx, float hy, float hz,
                                             float cc, float ss, bool seg) {
    float rx = px - cx, ry = py - cy, rz = pz - cz;
    float xr = rx * cc - ry * ss;
    float yr = rx * ss + ry * cc;
    bool in = (fabsf(xr) <= hx) && (fabsf(yr) <= hy) && (fabsf(rz) <= hz);
    return (in && seg) ? 1u : 0u;
}

template <bool VEC>
__global__ __launch_bounds__(256) void sgm_main(
    const float* __restrict__ points,   // [B][N][3]
    const float* __restrict__ segp,     // [B][N]
    const float* __restrict__ boxes,    // [B][MBOX][7]
    float* __restrict__ mask_out,       // [B][MBOX][N] (floats 0/1)
    unsigned int* __restrict__ cnt_ws,  // [B][MBOX]
    int N) {

    __shared__ float sbox[MBOX][8]; // cx,cy,cz,hx,hy,hz,cos(-yaw),sin(-yaw)
    __shared__ __attribute__((aligned(16))) unsigned char smask[8][MBOX][64]; // [wave][box][point]

    const int b   = blockIdx.y;
    const int tid = threadIdx.x;

    if (tid < MBOX) {
        const float* g = boxes + ((size_t)b * MBOX + tid) * 7;
        float yaw = g[6];
        sbox[tid][0] = g[0];
        sbox[tid][1] = g[1];
        sbox[tid][2] = g[2];
        sbox[tid][3] = g[3] * 0.5f;
        sbox[tid][4] = g[4] * 0.5f;
        sbox[tid][5] = g[5] * 0.5f;
        sbox[tid][6] =  cosf(yaw);   // cos(-yaw)
        sbox[tid][7] = -sinf(yaw);   // sin(-yaw)
    }
    __syncthreads();

    const int wave = tid >> 5;
    const int lane = tid & 31;
    const int wgl  = blockIdx.x * (blockDim.x >> 5) + wave;
    const int nw   = gridDim.x * (blockDim.x >> 5);
    const int numTiles = (N + 63) >> 6;

    const float* pbase = points + (size_t)b * N * 3;
    const float* sbase = segp + (size_t)b * N;
    float* mbase = mask_out + (size_t)b * MBOX * N;

    v8i acc[4];
#pragma unroll
    for (int g = 0; g < 4; ++g) acc[g] = (v8i){0, 0, 0, 0, 0, 0, 0, 0};

    const v8i bOnes = {0x01010101, 0x01010101, 0x01010101, 0x01010101,
                       0x01010101, 0x01010101, 0x01010101, 0x01010101};

    for (int t = wgl; t < numTiles; t += nw) {
        const int p0 = (t << 6) + (lane << 1);
        const bool full = ((t + 1) << 6) <= N; // wave-uniform: no divergence

        if (full) {
            // ---- guard-free fast path ----
            const float* p = pbase + (size_t)p0 * 3;
            const float px0 = __builtin_nontemporal_load(p + 0);
            const float py0 = __builtin_nontemporal_load(p + 1);
            const float pz0 = __builtin_nontemporal_load(p + 2);
            const float px1 = __builtin_nontemporal_load(p + 3);
            const float py1 = __builtin_nontemporal_load(p + 4);
            const float pz1 = __builtin_nontemporal_load(p + 5);
            const bool s0 = __builtin_nontemporal_load(sbase + p0)     > MOVING_THRESHOLD;
            const bool s1 = __builtin_nontemporal_load(sbase + p0 + 1) > MOVING_THRESHOLD;

#pragma unroll 4
            for (int m = 0; m < MBOX; ++m) {
                const float cx = sbox[m][0], cy = sbox[m][1], cz = sbox[m][2];
                const float hx = sbox[m][3], hy = sbox[m][4], hz = sbox[m][5];
                const float cc = sbox[m][6], ss = sbox[m][7];

                unsigned m0 = box_test(px0, py0, pz0, cx, cy, cz, hx, hy, hz, cc, ss, s0);
                unsigned m1 = box_test(px1, py1, pz1, cx, cy, cz, hx, hy, hz, cc, ss, s1);

                *(unsigned short*)&smask[wave][m][lane << 1] =
                    (unsigned short)(m0 | (m1 << 8));

                float* o = mbase + (size_t)m * N + p0;
                if constexpr (VEC) {
                    v2f v = {(float)m0, (float)m1}; // 8B-aligned: N even, p0 even
                    __builtin_nontemporal_store(v, (v2f*)o);
                } else {
                    __builtin_nontemporal_store((float)m0, o + 0);
                    __builtin_nontemporal_store((float)m1, o + 1);
                }
            }
        } else {
            // ---- tail tile (at most one per batch) ----
            const int p1 = p0 + 1;
            const bool ok0 = p0 < N;
            const bool ok1 = p1 < N;
            float px0 = 0.f, py0 = 0.f, pz0 = 0.f;
            float px1 = 0.f, py1 = 0.f, pz1 = 0.f;
            bool s0 = false, s1 = false;
            if (ok0) {
                const float* p = pbase + (size_t)p0 * 3;
                px0 = p[0]; py0 = p[1]; pz0 = p[2];
                s0 = sbase[p0] > MOVING_THRESHOLD;
            }
            if (ok1) {
                const float* p = pbase + (size_t)p1 * 3;
                px1 = p[0]; py1 = p[1]; pz1 = p[2];
                s1 = sbase[p1] > MOVING_THRESHOLD;
            }
#pragma unroll 4
            for (int m = 0; m < MBOX; ++m) {
                const float cx = sbox[m][0], cy = sbox[m][1], cz = sbox[m][2];
                const float hx = sbox[m][3], hy = sbox[m][4], hz = sbox[m][5];
                const float cc = sbox[m][6], ss = sbox[m][7];
                unsigned m0 = box_test(px0, py0, pz0, cx, cy, cz, hx, hy, hz, cc, ss, s0);
                unsigned m1 = box_test(px1, py1, pz1, cx, cy, cz, hx, hy, hz, cc, ss, s1);
                *(unsigned short*)&smask[wave][m][lane << 1] =
                    (unsigned short)(m0 | (m1 << 8));
                float* o = mbase + (size_t)m * N + p0;
                if (ok0) o[0] = (float)m0;
                if (ok1) o[1] = (float)m1;
            }
        }

        // Same-wave DS ordering: ensure staged bytes are readable.
        asm volatile("s_wait_dscnt 0x0" ::: "memory");

        // A-matrix (16x64 IU8) fragment per CDNA5 layout:
        // lanes 0-15: row=lane,   K = {0-7,16-23,32-39,48-55}
        // lanes 16-31: row=lane-16, K = {8-15,24-31,40-47,56-63}
        const int row = lane & 15;
        const int kb0 = (lane >> 4) << 3;
#pragma unroll
        for (int g = 0; g < 4; ++g) {
            const unsigned char* rp = &smask[wave][(g << 4) + row][0];
            unsigned long long q0 = *(const unsigned long long*)(rp + kb0);
            unsigned long long q1 = *(const unsigned long long*)(rp + kb0 + 16);
            unsigned long long q2 = *(const unsigned long long*)(rp + kb0 + 32);
            unsigned long long q3 = *(const unsigned long long*)(rp + kb0 + 48);
            v8i a;
            a[0] = (int)q0; a[1] = (int)(q0 >> 32);
            a[2] = (int)q1; a[3] = (int)(q1 >> 32);
            a[4] = (int)q2; a[5] = (int)(q2 >> 32);
            a[6] = (int)q3; a[7] = (int)(q3 >> 32);
            // D[m][n] = sum_k A[m][k] * 1 : row sums = per-box counts.
            acc[g] = __builtin_amdgcn_wmma_i32_16x16x64_iu8(
                false, a, false, bOnes, acc[g], false, false);
        }
        asm volatile("" ::: "memory");
    }

    // D layout: lane 0, VGPR r -> box r ; lane 16, VGPR r -> box r+8 (per 16-box group).
    unsigned int* cnt = cnt_ws + b * MBOX;
    if (lane == 0 || lane == 16) {
        const int half = (lane >> 4) << 3; // 0 or 8
#pragma unroll
        for (int g = 0; g < 4; ++g) {
#pragma unroll
            for (int r = 0; r < 8; ++r) {
                atomicAdd(&cnt[(g << 4) + half + r], (unsigned int)acc[g][r]);
            }
        }
    }
}

__global__ void sgm_finalize(const unsigned int* __restrict__ cnt,
                             const int* __restrict__ minp,
                             float* __restrict__ out, int BM) {
    int i = blockIdx.x * blockDim.x + threadIdx.x;
    if (i < BM) {
        float c = (float)cnt[i];
        out[i] = c;                                        // counts
        out[BM + i] = (c >= (float)(*minp)) ? 1.0f : 0.0f; // matched
    }
}

extern "C" void kernel_launch(void* const* d_in, const int* in_sizes, int n_in,
                              void* d_out, int out_size, void* d_ws, size_t ws_size,
                              hipStream_t stream) {
    const float* points = (const float*)d_in[0];
    const float* segp   = (const float*)d_in[1];
    const float* boxes  = (const float*)d_in[2];
    const int*   minp   = (const int*)d_in[3];

    const int B  = in_sizes[2] / (MBOX * 7); // gt_boxes is [B][64][7]
    const int N  = in_sizes[1] / B;          // seg_pred is [B][N]
    const int BM = B * MBOX;

    float* out = (float*)d_out;
    unsigned int* cnt_ws = (unsigned int*)d_ws;

    sgm_init_counts<<<(BM + 255) / 256, 256, 0, stream>>>(cnt_ws, BM);

    dim3 grid(128, B);
    if ((N & 1) == 0) {
        sgm_main<true><<<grid, 256, 0, stream>>>(points, segp, boxes,
                                                 out + 2 * (size_t)BM, cnt_ws, N);
    } else {
        sgm_main<false><<<grid, 256, 0, stream>>>(points, segp, boxes,
                                                  out + 2 * (size_t)BM, cnt_ws, N);
    }

    sgm_finalize<<<(BM + 255) / 256, 256, 0, stream>>>(cnt_ws, minp, out, BM);
}
